// MultiHeadAttention_67645734912295
// MI455X (gfx1250) — matrix operand order repaired and checked by reading, not compiled, not run on hardware
//
#include <hip/hip_runtime.h>

// ---------------------------------------------------------------------------
// MHA for gfx1250 (MI455X): one-pass f32->bf16 conversion, software-pipelined
// bf16 WMMA GEMMs, flash attention with workgroup-shared double-buffered K/V
// tiles in LDS. All matmuls on v_wmma_f32_16x16x32_bf16.
// ---------------------------------------------------------------------------

typedef __bf16 bf16x16 __attribute__((ext_vector_type(16)));
typedef __bf16 bf16x2  __attribute__((ext_vector_type(2)));
typedef float  f32x8   __attribute__((ext_vector_type(8)));

union BF16Vec {
    bf16x16 v;
    unsigned short u[16];
    uint4 q[2];
};

union US8 {
    unsigned short u[8];
    uint4 q;
};

__device__ __forceinline__ unsigned short f2bf(float f) {
    unsigned int b = __float_as_uint(f);
    b += 0x7FFFu + ((b >> 16) & 1u);           // round-to-nearest-even
    return (unsigned short)(b >> 16);
}

// Packed pair conversion: two f32 -> one dword of two bf16.
__device__ __forceinline__ unsigned int pk_bf16(float lo, float hi) {
#if __has_builtin(__builtin_amdgcn_cvt_pk_bf16_f32)
    union { bf16x2 v; unsigned int u; } r;
    r.v = __builtin_amdgcn_cvt_pk_bf16_f32(lo, hi);
    return r.u;
#else
    return (unsigned int)f2bf(lo) | ((unsigned int)f2bf(hi) << 16);
#endif
}

__device__ __forceinline__ float red16_max(float v) {
#pragma unroll
    for (int off = 8; off > 0; off >>= 1) v = fmaxf(v, __shfl_xor(v, off, 16));
    return v;
}
__device__ __forceinline__ float red16_sum(float v) {
#pragma unroll
    for (int off = 8; off > 0; off >>= 1) v += __shfl_xor(v, off, 16);
    return v;
}

#define WMMA_BF16(A, B, C) \
    __builtin_amdgcn_wmma_f32_16x16x32_bf16(false, (A), false, (B), (short)0, (C), false, false)

// ---------------------------------------------------------------------------
// f32 -> bf16 bulk conversion: 8 elements per thread, fully vectorized.
// ---------------------------------------------------------------------------
__global__ __launch_bounds__(256) void cvt_bf16_kernel(
    const float* __restrict__ src, unsigned short* __restrict__ dst, int n8)
{
    const int i = blockIdx.x * 256 + threadIdx.x;
    if (i >= n8) return;
    const float4 a = *(const float4*)(src + (size_t)i * 8);
    const float4 b = *(const float4*)(src + (size_t)i * 8 + 4);
    uint4 o;
    o.x = pk_bf16(a.x, a.y);
    o.y = pk_bf16(a.z, a.w);
    o.z = pk_bf16(b.x, b.y);
    o.w = pk_bf16(b.z, b.w);
    *(uint4*)(dst + (size_t)i * 8) = o;
}

// ---------------------------------------------------------------------------
// GEMM operand load helpers (bf16, pre-converted).
// A 16x32 layout: K = h*16 + e (16 contiguous halves).
// B 32x16 layout: lane n = N, K = h*16 + e (16 contiguous halves).
// ---------------------------------------------------------------------------
__device__ __forceinline__ void loadA(BF16Vec& a, const unsigned short* X,
                                      int aRow, int k0, int h) {
    const unsigned short* ap = X + (size_t)aRow * 768 + k0 + h * 16;
    a.q[0] = *(const uint4*)(ap);
    a.q[1] = *(const uint4*)(ap + 8);
}
__device__ __forceinline__ void loadB4(BF16Vec* b, const unsigned short* W,
                                       int colBase, int n, int k0, int h) {
#pragma unroll
    for (int t = 0; t < 4; ++t) {
        const unsigned short* bp = W + (size_t)(colBase + t * 16 + n) * 768 + k0 + h * 16;
        b[t].q[0] = *(const uint4*)(bp);
        b[t].q[1] = *(const uint4*)(bp + 8);
    }
}

// ---------------------------------------------------------------------------
// Projection: Out(bf16) = X(bf16)[8192x768] * W(bf16)[768x768]^T
// Software-pipelined: operands for chunk kb+1 are in flight during chunk kb.
// transposed==0: Out layout [B,H,S,64]; transposed==1: Out layout [B,H,64,S].
// ---------------------------------------------------------------------------
__global__ __launch_bounds__(256) void proj_bf16_kernel(
    const unsigned short* __restrict__ X, const unsigned short* __restrict__ W,
    unsigned short* __restrict__ Out, int transposed, float scale)
{
    const int lane = threadIdx.x & 31;
    const int wave = threadIdx.x >> 5;
    const int n = lane & 15;
    const int h = lane >> 4;
    const int task = blockIdx.x * 8 + wave;   // 6144 tasks
    const int rt = task / 12;                 // row tile (16 rows of 8192)
    const int cg = task % 12;                 // 64-col group
    const int row0 = rt * 16;
    const int aRow = row0 + n;
    const int colBase = cg * 64;

    f32x8 acc[4];
#pragma unroll
    for (int t = 0; t < 4; ++t)
#pragma unroll
        for (int r = 0; r < 8; ++r) acc[t][r] = 0.0f;

    BF16Vec a0, a1, b0[4], b1[4];
    loadA(a0, X, aRow, 0, h);
    loadB4(b0, W, colBase, n, 0, h);

    for (int kb = 0; kb < 24; kb += 2) {
        loadA(a1, X, aRow, (kb + 1) * 32, h);
        loadB4(b1, W, colBase, n, (kb + 1) * 32, h);
#pragma unroll
        for (int t = 0; t < 4; ++t) acc[t] = WMMA_BF16(a0.v, b0[t].v, acc[t]);

        const int k2 = (kb + 2 < 24) ? (kb + 2) * 32 : 0;   // tail reload harmless
        loadA(a0, X, aRow, k2, h);
        loadB4(b0, W, colBase, n, k2, h);
#pragma unroll
        for (int t = 0; t < 4; ++t) acc[t] = WMMA_BF16(a1.v, b1[t].v, acc[t]);
    }

    // C layout: lane holds N = n; reg r -> M = h*8 + r
#pragma unroll
    for (int t = 0; t < 4; ++t) {
        const int col = colBase + t * 16 + n;
        const int head = col >> 6;
        const int dcol = col & 63;
        if (!transposed) {
#pragma unroll
            for (int r = 0; r < 8; ++r) {
                const int row = row0 + h * 8 + r;
                const int bIdx = row >> 12;
                const int s = row & 4095;
                Out[(((size_t)(bIdx * 12 + head) * 4096 + s) << 6) + dcol] =
                    f2bf(acc[t][r] * scale);
            }
        } else {
            US8 tmp;
#pragma unroll
            for (int r = 0; r < 8; ++r) tmp.u[r] = f2bf(acc[t][r] * scale);
            const int row = row0 + h * 8;
            const int bIdx = row >> 12;
            const int s = row & 4095;                  // consecutive s for r=0..7
            *(uint4*)(Out + ((size_t)(bIdx * 12 + head) * 64 + dcol) * 4096 + s) = tmp.q;
        }
    }
}

// ---------------------------------------------------------------------------
// Flash attention. Workgroup = 8 waves sharing one (b,h); each wave owns 16
// query rows (128 per block). K/V 32-key chunks are cooperatively staged into
// double-buffered LDS tiles: global loads for chunk kb+1 issue first, compute
// for chunk kb runs on LDS, then staged data is stored and the block syncs.
// Qp,Kp: [B,H,S,64] bf16 (Q pre-scaled by 1/8). Vt: [B,H,64,S] bf16.
// AO: [B,S,768] bf16.
// ---------------------------------------------------------------------------
__global__ __launch_bounds__(256) void flash_attn_kernel(
    const unsigned short* __restrict__ Qp, const unsigned short* __restrict__ Kp,
    const unsigned short* __restrict__ Vt, unsigned short* __restrict__ AO)
{
    __shared__ __align__(16) unsigned short sK[2][32 * 64];   // 2 x 4 KB
    __shared__ __align__(16) unsigned short sV[2][64 * 32];   // 2 x 4 KB
    __shared__ __align__(16) unsigned short ldsP[8][512];     // per-wave P tile

    const int tid = threadIdx.x;
    const int lane = tid & 31;
    const int wave = tid >> 5;
    const int n = lane & 15;
    const int h = lane >> 4;
    const int bh = blockIdx.x >> 5;           // (b*12 + head), 0..23
    const int qb = blockIdx.x & 31;
    const int q0 = qb * 128 + wave * 16;      // query row base within S

    const unsigned short* Qh = Qp + (size_t)bh * 4096 * 64;
    const unsigned short* Kh = Kp + (size_t)bh * 4096 * 64;
    const unsigned short* Vh = Vt + (size_t)bh * 64 * 4096;

    // Cooperative staging pattern: 256 threads x one uint4 each per tile.
    const int krow = tid >> 3, kpart = tid & 7;        // K: 32 rows x 64 dims
    const int vrow = tid >> 2, vpart = tid & 3;        // V: 64 dims x 32 keys
    const unsigned short* kSrc = Kh + (size_t)krow * 64 + kpart * 8;
    const unsigned short* vSrc = Vh + (size_t)vrow * 4096 + vpart * 8;
    const int kOff = krow * 64 + kpart * 8;
    const int vOff = vrow * 32 + vpart * 8;

    // Q in A-layout, two K-chunks of 32 over head dim
    BF16Vec qa[2];
#pragma unroll
    for (int c = 0; c < 2; ++c) {
        const unsigned short* p = Qh + (size_t)(q0 + n) * 64 + c * 32 + h * 16;
        qa[c].q[0] = *(const uint4*)(p);
        qa[c].q[1] = *(const uint4*)(p + 8);
    }

    float m_i[8], l_i[8];
    f32x8 O[4];
#pragma unroll
    for (int r = 0; r < 8; ++r) { m_i[r] = -1e30f; l_i[r] = 0.0f; }
#pragma unroll
    for (int t = 0; t < 4; ++t)
#pragma unroll
        for (int r = 0; r < 8; ++r) O[t][r] = 0.0f;

    // Prime buffer 0 with chunk 0.
    {
        uint4 kr = *(const uint4*)(kSrc);
        uint4 vr = *(const uint4*)(vSrc);
        *(uint4*)&sK[0][kOff] = kr;
        *(uint4*)&sV[0][vOff] = vr;
    }
    __syncthreads();

    for (int kb = 0; kb < 128; ++kb) {
        const int cur = kb & 1;
        const int nxt = cur ^ 1;

        // 1) Issue global loads for chunk kb+1 (latency hidden by compute).
        const int k1 = (kb + 1 < 128) ? (kb + 1) * 32 : 0;
        const uint4 krn = *(const uint4*)(kSrc + (size_t)k1 * 64);
        const uint4 vrn = *(const uint4*)(vSrc + k1);

        // 2) Compute chunk kb from LDS.
        f32x8 sc[2];
#pragma unroll
        for (int kt = 0; kt < 2; ++kt) {
            f32x8 a;
#pragma unroll
            for (int r = 0; r < 8; ++r) a[r] = 0.0f;
#pragma unroll
            for (int c = 0; c < 2; ++c) {
                BF16Vec kv;
                const unsigned short* p = &sK[cur][(kt * 16 + n) * 64 + c * 32 + h * 16];
                kv.q[0] = *(const uint4*)(p);
                kv.q[1] = *(const uint4*)(p + 8);
                a = WMMA_BF16(qa[c].v, kv.v, a);
            }
            sc[kt] = a;
        }

        // Online softmax; row m = h*8 + r lives across 16 lanes of a half.
#pragma unroll
        for (int r = 0; r < 8; ++r) {
            float vmax = red16_max(fmaxf(sc[0][r], sc[1][r]));
            float mn = fmaxf(m_i[r], vmax);
            float p0 = __expf(sc[0][r] - mn);
            float p1 = __expf(sc[1][r] - mn);
            float al = __expf(m_i[r] - mn);
            float rs = red16_sum(p0 + p1);
            l_i[r] = l_i[r] * al + rs;
            m_i[r] = mn;
#pragma unroll
            for (int t = 0; t < 4; ++t) O[t][r] *= al;
            const int m = h * 8 + r;
            ldsP[wave][m * 32 + n] = f2bf(p0);
            ldsP[wave][m * 32 + 16 + n] = f2bf(p1);
        }

        // P in A-layout (per-wave region; DS ops in-order within a wave).
        BF16Vec pa;
        pa.q[0] = *(const uint4*)(&ldsP[wave][n * 32 + h * 16]);
        pa.q[1] = *(const uint4*)(&ldsP[wave][n * 32 + h * 16 + 8]);

#pragma unroll
        for (int t = 0; t < 4; ++t) {
            BF16Vec vv;
            const unsigned short* p = &sV[cur][(t * 16 + n) * 32 + h * 16];
            vv.q[0] = *(const uint4*)(p);
            vv.q[1] = *(const uint4*)(p + 8);
            O[t] = WMMA_BF16(pa.v, vv.v, O[t]);
        }

        // 3) Store staged chunk kb+1, 4) sync (also protects buffer reuse).
        *(uint4*)&sK[nxt][kOff] = krn;
        *(uint4*)&sV[nxt][vOff] = vrn;
        __syncthreads();
    }

    const int b = bh / 12, head = bh % 12;
#pragma unroll
    for (int r = 0; r < 8; ++r) {
        const float inv = 1.0f / l_i[r];
        const int s = q0 + h * 8 + r;
#pragma unroll
        for (int t = 0; t < 4; ++t) {
            const int col = head * 64 + t * 16 + n;
            AO[(size_t)(b * 4096 + s) * 768 + col] = f2bf(O[t][r] * inv);
        }
    }
}

// ---------------------------------------------------------------------------
// Output projection: out(f32) = AO(bf16)[8192x768] * Wo(bf16)[768x768]^T
// Same pipelined GEMM skeleton, f32 output.
// ---------------------------------------------------------------------------
__global__ __launch_bounds__(256) void out_proj_kernel(
    const unsigned short* __restrict__ X, const unsigned short* __restrict__ W,
    float* __restrict__ Out)
{
    const int lane = threadIdx.x & 31;
    const int wave = threadIdx.x >> 5;
    const int n = lane & 15;
    const int h = lane >> 4;
    const int task = blockIdx.x * 8 + wave;
    const int rt = task / 12;
    const int cg = task % 12;
    const int row0 = rt * 16;
    const int aRow = row0 + n;
    const int colBase = cg * 64;

    f32x8 acc[4];
#pragma unroll
    for (int t = 0; t < 4; ++t)
#pragma unroll
        for (int r = 0; r < 8; ++r) acc[t][r] = 0.0f;

    BF16Vec a0, a1, b0[4], b1[4];
    loadA(a0, X, aRow, 0, h);
    loadB4(b0, W, colBase, n, 0, h);

    for (int kb = 0; kb < 24; kb += 2) {
        loadA(a1, X, aRow, (kb + 1) * 32, h);
        loadB4(b1, W, colBase, n, (kb + 1) * 32, h);
#pragma unroll
        for (int t = 0; t < 4; ++t) acc[t] = WMMA_BF16(a0.v, b0[t].v, acc[t]);

        const int k2 = (kb + 2 < 24) ? (kb + 2) * 32 : 0;
        loadA(a0, X, aRow, k2, h);
        loadB4(b0, W, colBase, n, k2, h);
#pragma unroll
        for (int t = 0; t < 4; ++t) acc[t] = WMMA_BF16(a1.v, b1[t].v, acc[t]);
    }

#pragma unroll
    for (int t = 0; t < 4; ++t) {
        const int col = colBase + t * 16 + n;
#pragma unroll
        for (int r = 0; r < 8; ++r) {
            const int row = row0 + h * 8 + r;
            Out[(size_t)row * 768 + col] = acc[t][r];
        }
    }
}

// ---------------------------------------------------------------------------
extern "C" void kernel_launch(void* const* d_in, const int* in_sizes, int n_in,
                              void* d_out, int out_size, void* d_ws, size_t ws_size,
                              hipStream_t stream) {
    const float* q  = (const float*)d_in[0];
    const float* k  = (const float*)d_in[1];
    const float* v  = (const float*)d_in[2];
    const float* Wq = (const float*)d_in[3];
    const float* Wk = (const float*)d_in[4];
    const float* Wv = (const float*)d_in[5];
    const float* Wo = (const float*)d_in[6];
    float* out = (float*)d_out;

    char* ws = (char*)d_ws;
    const size_t SZ  = (size_t)2 * 12 * 4096 * 64 * sizeof(unsigned short); // 12.58 MB
    const size_t WSZ = (size_t)768 * 768 * sizeof(unsigned short);          // 1.18 MB
    unsigned short* Qp  = (unsigned short*)(ws);            // [B,H,S,64] bf16 (pre-scaled)
    unsigned short* Kp  = (unsigned short*)(ws + SZ);       // [B,H,S,64] bf16
    unsigned short* Vt  = (unsigned short*)(ws + 2 * SZ);   // [B,H,64,S] bf16
    unsigned short* AO  = (unsigned short*)(ws + 3 * SZ);   // [B,S,768]  bf16
    unsigned short* Xq  = (unsigned short*)(ws + 4 * SZ);   // [B,S,768]  bf16
    unsigned short* Xk  = (unsigned short*)(ws + 5 * SZ);
    unsigned short* Xv  = (unsigned short*)(ws + 6 * SZ);
    unsigned short* Wqb = (unsigned short*)(ws + 7 * SZ);
    unsigned short* Wkb = (unsigned short*)(ws + 7 * SZ + WSZ);
    unsigned short* Wvb = (unsigned short*)(ws + 7 * SZ + 2 * WSZ);
    unsigned short* Wob = (unsigned short*)(ws + 7 * SZ + 3 * WSZ);

    dim3 blk(256);
    const int nx8 = (2 * 4096 * 768) / 8;   // 786432 groups per activation
    const int nw8 = (768 * 768) / 8;        // 73728 groups per weight
    cvt_bf16_kernel<<<(nx8 + 255) / 256, blk, 0, stream>>>(q,  Xq,  nx8);
    cvt_bf16_kernel<<<(nx8 + 255) / 256, blk, 0, stream>>>(k,  Xk,  nx8);
    cvt_bf16_kernel<<<(nx8 + 255) / 256, blk, 0, stream>>>(v,  Xv,  nx8);
    cvt_bf16_kernel<<<(nw8 + 255) / 256, blk, 0, stream>>>(Wq, Wqb, nw8);
    cvt_bf16_kernel<<<(nw8 + 255) / 256, blk, 0, stream>>>(Wk, Wkb, nw8);
    cvt_bf16_kernel<<<(nw8 + 255) / 256, blk, 0, stream>>>(Wv, Wvb, nw8);
    cvt_bf16_kernel<<<(nw8 + 255) / 256, blk, 0, stream>>>(Wo, Wob, nw8);

    proj_bf16_kernel<<<768, blk, 0, stream>>>(Xq, Wqb, Qp, 0, 0.125f); // 1/sqrt(64) folded
    proj_bf16_kernel<<<768, blk, 0, stream>>>(Xk, Wkb, Kp, 0, 1.0f);
    proj_bf16_kernel<<<768, blk, 0, stream>>>(Xv, Wvb, Vt, 1, 1.0f);
    flash_attn_kernel<<<768, blk, 0, stream>>>(Qp, Kp, Vt, AO);
    out_proj_kernel<<<768, blk, 0, stream>>>(AO, Wob, out);
}
